// Encoder_21758304322267
// MI455X (gfx1250) — matrix operand order repaired and checked by reading, not compiled, not run on hardware
//
#include <hip/hip_runtime.h>
#include <hip/hip_bf16.h>

// ---------------- problem constants ----------------
#define LS   2
#define EE   512
#define HH   1024
#define BB   64
#define TT   128
#define NG   3072          // 3*H
#define LN_EPS 1e-5f

#define NWG  24            // workgroups per direction in persistent GRU kernel
#define NT_PER_WG 8        // 192 n-tiles / 24 WGs

typedef __attribute__((ext_vector_type(16))) __bf16 v16bf;
typedef __attribute__((ext_vector_type(8)))  float  v8f;

// ---------------- helpers ----------------
__device__ __forceinline__ unsigned short f2bf(float f) {
  unsigned u = __builtin_bit_cast(unsigned, f);
  u += 0x7FFFu + ((u >> 16) & 1u);        // round-to-nearest-even
  return (unsigned short)(u >> 16);
}

// 16-bit A-matrix 16x32 layout used here (contiguous k = 16*(lane/16) + v): row-in-tile mr (0..15), k-in-tile kk (0..31)
// -> (lane, value-slot)
__device__ __forceinline__ void a_lane_v(int mr, int kk, int& ln, int& v) {
  if (kk < 16)      { ln = mr;      v = kk;      }
  else              { ln = mr + 16; v = kk - 16; }
}

__device__ __forceinline__ v8f wmma_bf16p(const unsigned short* a,
                                          const unsigned short* b, v8f c) {
  v16bf av = *(const v16bf*)a;
  v16bf bv = *(const v16bf*)b;
  return __builtin_amdgcn_wmma_f32_16x16x32_bf16(false, av, false, bv,
                                                 (short)0, c, false, false);
}

__device__ __forceinline__ float wave_red_sum(float v) {
#pragma unroll
  for (int m = 16; m; m >>= 1) v += __shfl_xor(v, m, 32);
  return v;   // butterfly: all 32 lanes hold the sum
}

// gfx1250 async copy global -> LDS (per lane: 16 bytes), tracked by ASYNCcnt
__device__ __forceinline__ void async_copy_b128(unsigned lds_off,
                                                const unsigned short* gsrc) {
  asm volatile("global_load_async_to_lds_b128 %0, %1, off"
               :: "v"(lds_off), "v"(gsrc)
               : "memory");
}
__device__ __forceinline__ void wait_async0() {
  asm volatile("s_wait_asynccnt 0x0" ::: "memory");
}

__device__ __forceinline__ void grid_barrier(unsigned int* cnt, unsigned int target) {
  __syncthreads();
  if (threadIdx.x == 0) {
    __threadfence();                       // release h / ah writes (device scope)
    atomicAdd(cnt, 1u);
    while (__atomic_load_n(cnt, __ATOMIC_RELAXED) < target)
      __builtin_amdgcn_s_sleep(1);
  }
  __syncthreads();
  __threadfence();                         // acquire other WGs' writes
}

// ---------------- weight / input tiling kernels ----------------
// Wh (2,LS,H,3H) f32 -> bf16 B-tiles  [dl(4)][nt(192)][kt(32)][lane(32)][v(16)]
__global__ void k_tile_wh(const float* __restrict__ Wh, unsigned short* __restrict__ dst) {
  size_t i = (size_t)blockIdx.x * 256 + threadIdx.x;
  if (i >= (size_t)4 * 192 * 32 * 512) return;
  int v = i & 15; size_t t1 = i >> 4;
  int lane = t1 & 31; size_t t2 = t1 >> 5;
  int kt = t2 & 31; size_t t3 = t2 >> 5;
  int nt = (int)(t3 % 192); int dl = (int)(t3 / 192);
  int K = kt * 32 + ((lane < 16) ? v : v + 16);
  int N = nt * 16 + (lane & 15);
  dst[i] = f2bf(Wh[((size_t)dl * HH + K) * NG + N]);
}

// Wx layer0 only (2,LS,E,3H) -> [dir(2)][nt(192)][kt(16)][lane][v]
__global__ void k_tile_wx(const float* __restrict__ Wx, unsigned short* __restrict__ dst) {
  size_t i = (size_t)blockIdx.x * 256 + threadIdx.x;
  if (i >= (size_t)2 * 192 * 16 * 512) return;
  int v = i & 15; size_t t1 = i >> 4;
  int lane = t1 & 31; size_t t2 = t1 >> 5;
  int kt = t2 & 15; size_t t3 = t2 >> 4;
  int nt = (int)(t3 % 192); int dir = (int)(t3 / 192);
  int K = kt * 32 + ((lane < 16) ? v : v + 16);
  int N = nt * 16 + (lane & 15);
  dst[i] = f2bf(Wx[((size_t)(dir * LS) * EE + K) * NG + N]);
}

// Wo (2H,H) -> [nt(64)][kt(64)][lane][v]
__global__ void k_tile_wo(const float* __restrict__ Wo, unsigned short* __restrict__ dst) {
  size_t i = (size_t)blockIdx.x * 256 + threadIdx.x;
  if (i >= (size_t)64 * 64 * 512) return;
  int v = i & 15; size_t t1 = i >> 4;
  int lane = t1 & 31; size_t t2 = t1 >> 5;
  int kt = t2 & 63; int nt = (int)(t2 >> 6);
  int K = kt * 32 + ((lane < 16) ? v : v + 16);
  int N = nt * 16 + (lane & 15);
  dst[i] = f2bf(Wo[(size_t)K * HH + N]);
}

// embedding gather -> xs A-tiles [mtile(512)][kt(16)][lane][v], row = t*B+b
__global__ void k_embed(const int* __restrict__ inputs, const float* __restrict__ emb,
                        unsigned short* __restrict__ dst) {
  size_t i = (size_t)blockIdx.x * 256 + threadIdx.x;
  if (i >= (size_t)8192 * 512) return;
  int row = (int)(i >> 9); int k = (int)(i & 511);
  int t = row >> 6, b = row & 63;
  int tok = inputs[b * TT + t];
  float val = emb[(size_t)tok * EE + k];
  int mt = row >> 4, mr = row & 15, kt = k >> 5, kk = k & 31;
  int ln, v; a_lane_v(mr, kk, ln, v);
  dst[(((size_t)mt * 16 + kt) * 32 + ln) * 16 + v] = f2bf(val);
}

// ---------------- big parallel GEMM: ax = xs @ Wx + bx  (raw, LN after) ----------------
// block: one (dir, m-tile, 8 consecutive n-tiles); A staged via async copy to LDS.
__global__ void __launch_bounds__(256)
k_gemm_ax(const unsigned short* __restrict__ xsT,
          const unsigned short* __restrict__ wxT,
          const float* __restrict__ bx, float* __restrict__ ax) {
  __shared__ unsigned short sA[16 * 512];               // 16 KB: [kt(16)][512]
  int id = blockIdx.x;                                  // 2*512*24
  int dir = id / (512 * 24);
  int rem = id % (512 * 24);
  int mt = rem / 24, ntg = rem % 24;
  int wave = threadIdx.x >> 5, lane = threadIdx.x & 31;
  int nt = ntg * 8 + wave;

  const unsigned lb = (unsigned)(size_t)(void*)sA;
  const unsigned short* src = xsT + (size_t)mt * 16 * 512;
  for (int i = threadIdx.x; i < 1024; i += 256)         // 16KB in b128 chunks
    async_copy_b128(lb + i * 16, src + i * 8);
  wait_async0();
  __syncthreads();

  const unsigned short* Bp = wxT + (size_t)(dir * 192 + nt) * 16 * 512;
  v8f acc = {};
#pragma unroll 4
  for (int kt = 0; kt < 16; ++kt)
    acc = wmma_bf16p(sA + (size_t)kt * 512 + lane * 16,
                     Bp + (size_t)kt * 512 + lane * 16, acc);
  int col = nt * 16 + (lane & 15);
  int rb  = (lane < 16) ? 0 : 8;
  float bxi = bx[(size_t)(dir * LS) * NG + col];
#pragma unroll
  for (int r = 0; r < 8; ++r)
    ax[((size_t)dir * 8192 + mt * 16 + rb + r) * NG + col] = acc[r] + bxi;
}

// in-place LayerNorm+scale over each (dir,row,gate) chunk of 1024
__global__ void k_ln_ax(const float* __restrict__ gxg, const float* __restrict__ gxb,
                        float* __restrict__ ax) {
  int wid = blockIdx.x * 8 + (threadIdx.x >> 5);        // 2*8192*3 pairs
  if (wid >= 2 * 8192 * 3) return;
  int lane = threadIdx.x & 31;
  int dir = wid / (8192 * 3);
  int rem = wid % (8192 * 3);
  int row = rem / 3, g = rem % 3;
  float* p = ax + ((size_t)dir * 8192 + row) * NG + g * 1024;
  float s1 = 0.f, s2 = 0.f;
  for (int c = lane; c < 1024; c += 32) { float v = p[c]; s1 += v; s2 += v * v; }
  s1 = wave_red_sum(s1); s2 = wave_red_sum(s2);
  float mu = s1 * (1.f / 1024.f);
  float rs = rsqrtf(s2 * (1.f / 1024.f) - mu * mu + LN_EPS);
  const float* gg = gxg + (size_t)(dir * LS) * NG + g * 1024;
  const float* gb = gxb + (size_t)(dir * LS) * NG + g * 1024;
  for (int c = lane; c < 1024; c += 32)
    p[c] = (p[c] - mu) * rs * gg[c] + gb[c];
}

// constant ax for layers >= 1 (input is zero): LN(bx[dir][1]) * gxg + gxb
__global__ void k_axc1(const float* __restrict__ bx, const float* __restrict__ gxg,
                       const float* __restrict__ gxb, float* __restrict__ axc1) {
  int wid = threadIdx.x >> 5;
  if (wid >= 6) return;
  int lane = threadIdx.x & 31;
  int dir = wid / 3, g = wid % 3;
  const float* p = bx + (size_t)(dir * LS + 1) * NG + g * 1024;
  float s1 = 0.f, s2 = 0.f;
  for (int c = lane; c < 1024; c += 32) { float v = p[c]; s1 += v; s2 += v * v; }
  s1 = wave_red_sum(s1); s2 = wave_red_sum(s2);
  float mu = s1 * (1.f / 1024.f);
  float rs = rsqrtf(s2 * (1.f / 1024.f) - mu * mu + LN_EPS);
  const float* gg = gxg + (size_t)(dir * LS + 1) * NG + g * 1024;
  const float* gb = gxb + (size_t)(dir * LS + 1) * NG + g * 1024;
  for (int c = lane; c < 1024; c += 32)
    axc1[(size_t)dir * NG + g * 1024 + c] = (p[c] - mu) * rs * gg[c] + gb[c];
}

__global__ void k_init(float* __restrict__ h, unsigned short* __restrict__ hbT,
                       unsigned int* __restrict__ cnt) {
  int gtid = blockIdx.x * 256 + threadIdx.x;
  if (gtid < 131072) { h[gtid] = 0.f; hbT[gtid] = 0; }
  if (gtid < 2) cnt[gtid] = 0u;
}

// ---------------- persistent recurrent kernel ----------------
// hbT layout: [dir][kt(32)][mt(4)][lane(32)][v(16)] -> each 16-kt half is contiguous 64KB
__global__ void __launch_bounds__(256)
k_gru(const unsigned short* __restrict__ whT, const float* __restrict__ bh,
      const float* __restrict__ ghg, const float* __restrict__ ghb,
      const float* __restrict__ ax, const float* __restrict__ axc1,
      float* __restrict__ ah, float* __restrict__ stats,
      float* __restrict__ h, unsigned short* __restrict__ hbT,
      unsigned int* __restrict__ cnt, float* __restrict__ out_ctx) {
  __shared__ unsigned short sA[16 * 4 * 512];           // 64 KB: [ktLocal(16)][mt(4)][512]
  const int dir  = blockIdx.x / NWG;
  const int wg   = blockIdx.x % NWG;
  const int lane = threadIdx.x & 31;
  const int wave = threadIdx.x >> 5;
  const unsigned lb = (unsigned)(size_t)(void*)sA;

  unsigned int* cnt_d = cnt + dir;
  float* ah_d    = ah    + (size_t)dir * BB * NG;
  float* stats_d = stats + (size_t)dir * 192 * 2;
  float* h_d     = h     + (size_t)dir * BB * HH;
  unsigned short* hbT_d = hbT + (size_t)dir * 32 * 4 * 512;
  const float* ax_d = ax + (size_t)dir * 8192 * NG;

  unsigned int epoch = 0;
  for (int s = 0; s < TT; ++s) {
    for (int l = 0; l < LS; ++l) {
      // -------- phase A: ah = h_bf16 @ Wh (A staged in LDS via async copies) --------
      {
        const int nt = wg * NT_PER_WG + wave;            // 0..191
        const unsigned short* Wt = whT + (size_t)((dir * LS + l) * 192 + nt) * 32 * 512;
        v8f acc[4] = {};
        for (int half = 0; half < 2; ++half) {
          const unsigned short* srcp = hbT_d + (size_t)half * 16 * 4 * 512;
          for (int i = threadIdx.x; i < 4096; i += 256)  // 64KB in b128 chunks
            async_copy_b128(lb + i * 16, srcp + i * 8);
          wait_async0();
          __syncthreads();
          for (int kt = 0; kt < 16; ++kt) {
            const int ktg = half * 16 + kt;
            const unsigned short* bp = Wt + (size_t)ktg * 512 + lane * 16;
            if (ktg + 1 < 32)
              __builtin_prefetch(Wt + (size_t)(ktg + 1) * 512 + lane * 16, 0, 0);
            v16bf bv = *(const v16bf*)bp;
#pragma unroll
            for (int mt = 0; mt < 4; ++mt) {
              v16bf av = *(const v16bf*)(sA + (size_t)(kt * 4 + mt) * 512 + lane * 16);
              acc[mt] = __builtin_amdgcn_wmma_f32_16x16x32_bf16(
                  false, av, false, bv, (short)0, acc[mt], false, false);
            }
          }
          __syncthreads();                               // before overwriting sA
        }
        const int col = nt * 16 + (lane & 15);
        const int rb  = (lane < 16) ? 0 : 8;
#pragma unroll
        for (int mt = 0; mt < 4; ++mt)
#pragma unroll
          for (int r = 0; r < 8; ++r)
            ah_d[(size_t)(mt * 16 + rb + r) * NG + col] = acc[mt][r];
      }
      grid_barrier(cnt_d, (++epoch) * NWG);

      // -------- LN stats: one wave per (row,gate) over 1024 cols ---------------------
      {
        const int pair = wg * 8 + wave;                  // 0..191
        const int row = pair / 3, g = pair % 3;
        const float* bhp = bh + (size_t)(dir * LS + l) * NG + g * 1024;
        const float* ap  = ah_d + (size_t)row * NG + g * 1024;
        float s1 = 0.f, s2 = 0.f;
        for (int c = lane; c < 1024; c += 32) { float v = ap[c] + bhp[c]; s1 += v; s2 += v * v; }
        s1 = wave_red_sum(s1); s2 = wave_red_sum(s2);
        if (lane == 0) {
          float mu = s1 * (1.f / 1024.f);
          stats_d[pair * 2]     = mu;
          stats_d[pair * 2 + 1] = rsqrtf(s2 * (1.f / 1024.f) - mu * mu + LN_EPS);
        }
      }
      grid_barrier(cnt_d, (++epoch) * NWG);

      // -------- gates + h update + ctx write -----------------------------------------
      {
        const int gtid = wg * 256 + threadIdx.x;         // 0..6143
        const float* ghg_p = ghg + (size_t)(dir * LS + l) * NG;
        const float* ghb_p = ghb + (size_t)(dir * LS + l) * NG;
        const float* bh_p  = bh  + (size_t)(dir * LS + l) * NG;
        const int tt = dir ? (TT - 1 - s) : s;
        const float* ax_p = ax_d + (size_t)tt * BB * NG;
        for (int idx = gtid; idx < BB * HH; idx += NWG * 256) {
          int row = idx >> 10, col = idx & 1023;
          float hv[3], av[3];
#pragma unroll
          for (int g = 0; g < 3; ++g) {
            int j = g * 1024 + col;
            float mu = stats_d[(row * 3 + g) * 2];
            float rs = stats_d[(row * 3 + g) * 2 + 1];
            float a = ah_d[(size_t)row * NG + j] + bh_p[j];
            hv[g] = (a - mu) * rs * ghg_p[j] + ghb_p[j];
            av[g] = (l == 0) ? ax_p[(size_t)row * NG + j] : axc1[(size_t)dir * NG + j];
          }
          float r = 1.f / (1.f + __expf(-(av[0] + hv[0])));
          float z = 1.f / (1.f + __expf(-(av[1] + hv[1])));
          float n = tanhf(av[2] + r * hv[2]);
          float hn = (1.f - z) * n + z * h_d[idx];
          h_d[idx] = hn;
          int mt = row >> 4, mr = row & 15, kt = col >> 5, kk = col & 31;
          int ln, v; a_lane_v(mr, kk, ln, v);
          hbT_d[(((size_t)kt * 4 + mt) * 512) + ln * 16 + v] = f2bf(hn);
          if (l == LS - 1)
            out_ctx[((size_t)tt * BB + row) * (2 * HH) + (size_t)dir * HH + col] = hn;
        }
      }
      grid_barrier(cnt_d, (++epoch) * NWG);
    }
  }
}

// ---------------- final projection ----------------
__global__ void k_build_hcat(const float* __restrict__ h, unsigned short* __restrict__ dst) {
  int i = blockIdx.x * 256 + threadIdx.x;
  if (i >= BB * 2 * HH) return;
  int row = i >> 11, k = i & 2047;
  float val = (k < HH) ? h[(size_t)row * HH + k]
                       : h[(size_t)BB * HH + (size_t)row * HH + (k - HH)];
  int mt = row >> 4, mr = row & 15, kt = k >> 5, kk = k & 31;
  int ln, v; a_lane_v(mr, kk, ln, v);
  dst[(((size_t)mt * 64 + kt) * 32 + ln) * 16 + v] = f2bf(val);
}

__global__ void __launch_bounds__(256)
k_gemm_out(const unsigned short* __restrict__ hcT,
           const unsigned short* __restrict__ woT,
           const float* __restrict__ bo, float* __restrict__ hid) {
  __shared__ unsigned short sA[64 * 512];               // 64 KB: [kt(64)][512]
  int wave = threadIdx.x >> 5, lane = threadIdx.x & 31;
  int mt = blockIdx.x / 8, ntg = blockIdx.x % 8;
  int nt = ntg * 8 + wave;

  const unsigned lb = (unsigned)(size_t)(void*)sA;
  const unsigned short* src = hcT + (size_t)mt * 64 * 512;
  for (int i = threadIdx.x; i < 4096; i += 256)
    async_copy_b128(lb + i * 16, src + i * 8);
  wait_async0();
  __syncthreads();

  v8f acc = {};
#pragma unroll 4
  for (int kt = 0; kt < 64; ++kt)
    acc = wmma_bf16p(sA + (size_t)kt * 512 + lane * 16,
                     woT + (size_t)(nt * 64 + kt) * 512 + lane * 16, acc);
  int col = nt * 16 + (lane & 15);
  int rb  = (lane < 16) ? 0 : 8;
  float b = bo[col];
#pragma unroll
  for (int r = 0; r < 8; ++r)
    hid[(size_t)(mt * 16 + rb + r) * HH + col] = tanhf(acc[r] + b);
}

// ---------------- host launch ----------------
extern "C" void kernel_launch(void* const* d_in, const int* in_sizes, int n_in,
                              void* d_out, int out_size, void* d_ws, size_t ws_size,
                              hipStream_t stream) {
  const int*   inputs = (const int*)  d_in[0];
  const float* emb    = (const float*)d_in[2];
  const float* Wx     = (const float*)d_in[3];
  const float* Wh     = (const float*)d_in[4];
  const float* bx     = (const float*)d_in[5];
  const float* bh     = (const float*)d_in[6];
  const float* gxg    = (const float*)d_in[7];
  const float* gxb    = (const float*)d_in[8];
  const float* ghg    = (const float*)d_in[9];
  const float* ghb    = (const float*)d_in[10];
  const float* Wo     = (const float*)d_in[11];
  const float* bo     = (const float*)d_in[12];
  float* out = (float*)d_out;

  char* ws = (char*)d_ws;
  size_t off = 0;
  auto alloc = [&](size_t bytes) {
    void* p = ws + off;
    off = (off + bytes + 255) & ~(size_t)255;
    return p;
  };
  // workspace layout (~248 MB total)
  unsigned short* whT  = (unsigned short*)alloc((size_t)4 * 192 * 32 * 512 * 2);
  unsigned short* wxT  = (unsigned short*)alloc((size_t)2 * 192 * 16 * 512 * 2);
  unsigned short* woT  = (unsigned short*)alloc((size_t)64 * 64 * 512 * 2);
  unsigned short* xsT  = (unsigned short*)alloc((size_t)512 * 16 * 512 * 2);
  float* ax    = (float*)alloc((size_t)2 * 8192 * NG * 4);
  float* ah    = (float*)alloc((size_t)2 * BB * NG * 4);
  float* hbuf  = (float*)alloc((size_t)2 * BB * HH * 4);
  float* axc1  = (float*)alloc((size_t)2 * NG * 4);
  float* stats = (float*)alloc((size_t)2 * 192 * 2 * 4);
  unsigned short* hbT = (unsigned short*)alloc((size_t)2 * 32 * 4 * 512 * 2);
  unsigned short* hcT = (unsigned short*)alloc((size_t)4 * 64 * 512 * 2);
  unsigned int* cnt   = (unsigned int*)alloc(256);

  dim3 blk(256);
  k_tile_wh <<<49152, blk, 0, stream>>>(Wh, whT);
  k_tile_wx <<<12288, blk, 0, stream>>>(Wx, wxT);
  k_tile_wo <<< 8192, blk, 0, stream>>>(Wo, woT);
  k_embed   <<<16384, blk, 0, stream>>>(inputs, emb, xsT);
  k_gemm_ax <<<24576, blk, 0, stream>>>(xsT, wxT, bx, ax);
  k_ln_ax   <<< 6144, blk, 0, stream>>>(gxg, gxb, ax);
  k_init    <<<  512, blk, 0, stream>>>(hbuf, hbT, cnt);
  k_axc1    <<<    1, blk, 0, stream>>>(bx, gxg, gxb, axc1);
  k_gru     <<<2 * NWG, blk, 0, stream>>>(whT, bh, ghg, ghb, ax, axc1, ah, stats,
                                          hbuf, hbT, cnt, out);
  k_build_hcat<<<512, blk, 0, stream>>>(hbuf, hcT);
  k_gemm_out  <<< 32, blk, 0, stream>>>(hcT, woT, bo, out + (size_t)TT * BB * 2 * HH);
}
